// Encoder_53523882443679
// MI455X (gfx1250) — matrix operand order repaired and checked by reading, not compiled, not run on hardware
//
#include <hip/hip_runtime.h>
#include <hip/hip_bf16.h>

// ---------------------------------------------------------------------------
// GRU (B=64, T=512, F=64, H=256) + dense head, for gfx1250 (CDNA5, wave32).
//
//  * Weights AND the x input are pre-packed (one-time kernels) into the exact
//    V_WMMA_F32_16X16X32_BF16 lane layouts, so the persistent scan loop does
//    nothing but b128 loads + WMMA + gate VALU.
//  * Persistent GRU kernel: 4 workgroups (16-row batch tiles), 8 waves each.
//    h lives in LDS (fp32 + bf16 mirror). Per step, per wave: two column
//    groups of 16, four f32 accumulators each (z, r, xh, rh kept separate so
//    hh = tanh(xh + r*rh) matches the reference dataflow), 60 WMMAs.
//  * An opaque SGPR offset (empty asm) is added to the weight base pointers
//    inside the loop so LICM cannot hoist the (t-invariant) fragment loads
//    and spill them to scratch, while the loads keep their GLOBAL address
//    space (global_load_b128, LOADcnt-only tracking).
//  * Dense head: 16 waves, WMMA over K=256, tanh on accumulators.
// ---------------------------------------------------------------------------

typedef __attribute__((ext_vector_type(16))) __bf16 v16bf;
typedef __attribute__((ext_vector_type(8)))  float  v8f;

#define GRU_B 64
#define GRU_T 512
#define GRU_F 64
#define GRU_H 256
#define GRU_G (3 * GRU_H)   // 768

#define KT_X  (GRU_F / 32)   // 2
#define KT_R  (GRU_H / 32)   // 8
#define NT_G  (GRU_G / 16)   // 48
#define NT_D  (64 / 16)      // 4
#define BLK_ELEMS 512        // 32 lanes * 16 bf16 = one 32x16 tile (1 KB)

union AFrag {
    unsigned short us[16];
    uint4          q[2];
    v16bf          v;
};

__device__ __forceinline__ unsigned short f2bf(float f) {
    unsigned int u = __float_as_uint(f);
    u += 0x7FFFu + ((u >> 16) & 1u);      // round to nearest even
    return (unsigned short)(u >> 16);
}

__device__ __forceinline__ float sigmoid_f(float x) {
    return 1.0f / (1.0f + __expf(-x));
}
__device__ __forceinline__ float tanh_f(float x) {
    float e = __expf(2.0f * x);
    return (e - 1.0f) / (e + 1.0f);
}

// Load one pre-packed fragment (32 lanes x 32 contiguous bytes).
__device__ __forceinline__ v16bf load_frag(const unsigned short* W,
                                           int blk, int lane) {
    const uint4* p = (const uint4*)(W + (size_t)blk * BLK_ELEMS + lane * 16);
    AFrag f;
    f.q[0] = p[0];
    f.q[1] = p[1];
    return f.v;
}

// Build an A fragment (16x32 bf16) from a row-major fp32 source row.
// A layout: k = e + 8*((lane>>4) + (e>>3)), row M = lane&15.
__device__ __forceinline__ v16bf load_a_f32(const float* __restrict__ row,
                                            int kbase, int lane) {
    AFrag f;
    int hi = (lane >> 4) << 3;
#pragma unroll
    for (int g = 0; g < 2; ++g) {
        int k0 = kbase + g * 16 + hi;
        float4 a = *(const float4*)(row + k0);
        float4 b = *(const float4*)(row + k0 + 4);
        f.us[g * 8 + 0] = f2bf(a.x); f.us[g * 8 + 1] = f2bf(a.y);
        f.us[g * 8 + 2] = f2bf(a.z); f.us[g * 8 + 3] = f2bf(a.w);
        f.us[g * 8 + 4] = f2bf(b.x); f.us[g * 8 + 5] = f2bf(b.y);
        f.us[g * 8 + 6] = f2bf(b.z); f.us[g * 8 + 7] = f2bf(b.w);
    }
    return f.v;
}

__device__ __forceinline__ v8f wmma_bf16(v16bf a, v16bf b, v8f c) {
    return __builtin_amdgcn_wmma_f32_16x16x32_bf16(
        false, a, false, b, (short)0, c, false, false);
}

// ---------------------------------------------------------------------------
// Pack a K x N fp32 row-major matrix into bf16 WMMA-B tiles.
// Block blk = nt*KT + kt. B layout: K = kt*32 + e + 8*((lane>>4) + (e>>3)),
// column n = nt*16 + (lane&15).
// ---------------------------------------------------------------------------
__global__ void pack_b_kernel(const float* __restrict__ src,
                              unsigned short* __restrict__ dst,
                              int K, int N) {
    int KT = K >> 5, NT = N >> 4;
    int tid = blockIdx.x * blockDim.x + threadIdx.x;
    if (tid >= KT * NT * 32) return;
    int lane = tid & 31;
    int blk  = tid >> 5;
    int kt = blk % KT;
    int nt = blk / KT;
    int n  = nt * 16 + (lane & 15);
    int kb = kt * 32 + ((lane >> 4) << 3);
    AFrag f;
#pragma unroll
    for (int e = 0; e < 16; ++e)
        f.us[e] = f2bf(src[(size_t)(kb + e + ((e >> 3) << 3)) * N + n]);
    uint4* d = (uint4*)(dst + (size_t)blk * BLK_ELEMS + lane * 16);
    d[0] = f.q[0];
    d[1] = f.q[1];
}

// ---------------------------------------------------------------------------
// Pack x (B,T,F) fp32 into A-fragment bf16 layout:
// block blk = (bt*T + t)*KT_X + kt, bt = batch tile of 16 rows.
// ---------------------------------------------------------------------------
__global__ void pack_a_x_kernel(const float* __restrict__ x,
                                unsigned short* __restrict__ Xp) {
    int tid = blockIdx.x * blockDim.x + threadIdx.x;
    int lane = tid & 31;
    int blk  = tid >> 5;
    if (blk >= 4 * GRU_T * KT_X) return;
    int kt = blk % KT_X;
    int tt = (blk / KT_X) % GRU_T;
    int bt = (blk / KT_X) / GRU_T;
    const float* row =
        x + ((size_t)(bt * 16 + (lane & 15)) * GRU_T + tt) * GRU_F;
    AFrag f;
    f.v = load_a_f32(row, kt * 32, lane);
    uint4* d = (uint4*)(Xp + (size_t)blk * BLK_ELEMS + lane * 16);
    d[0] = f.q[0];
    d[1] = f.q[1];
}

// ---------------------------------------------------------------------------
// Persistent GRU scan kernel: grid = 4 blocks (batch tiles of 16), 256 thr.
// ---------------------------------------------------------------------------
__global__ __launch_bounds__(256, 1)
void gru_scan_kernel(const unsigned short* __restrict__ Xp,  // packed x frags
                     const float* __restrict__ bias,
                     const unsigned short* __restrict__ Wx,  // packed F x 3H
                     const unsigned short* __restrict__ Wr,  // packed H x 3H
                     float* __restrict__ outputs,            // (B,T,H)
                     float* __restrict__ hlast) {            // (B,H)
    __shared__ float          hF[16 * GRU_H];   // fp32 hidden state
    __shared__ unsigned short hB[16 * GRU_H];   // bf16 mirror for WMMA A

    const int wave = threadIdx.x >> 5;          // 0..7
    const int lane = threadIdx.x & 31;
    const int b0   = blockIdx.x * 16;
    const int M    = lane & 15;
    const int hi8  = (lane >> 4) << 3;

    for (int i = threadIdx.x; i < 16 * GRU_H; i += 256) {
        hF[i] = 0.0f;
        hB[i] = 0;
    }

    // t-invariant per-lane state
    int   jcol[2];
    float bz[2], br[2], bh[2];
#pragma unroll
    for (int cg = 0; cg < 2; ++cg) {
        jcol[cg] = (wave + 8 * cg) * 16 + (lane & 15);
        bz[cg] = bias[jcol[cg]];
        br[cg] = bias[GRU_H + jcol[cg]];
        bh[cg] = bias[2 * GRU_H + jcol[cg]];
    }
    const unsigned short* xbase =
        Xp + (size_t)blockIdx.x * GRU_T * KT_X * BLK_ELEMS;
    __syncthreads();

    for (int t = 0; t < GRU_T; ++t) {
        const unsigned short* xblk = xbase + (size_t)t * KT_X * BLK_ELEMS;
        if (t + 1 < GRU_T)
            __builtin_prefetch(xblk + KT_X * BLK_ELEMS, 0, 0);

        AFrag ax[KT_X];
#pragma unroll
        for (int kt = 0; kt < KT_X; ++kt) {
            const uint4* p = (const uint4*)(xblk + kt * BLK_ELEMS + lane * 16);
            ax[kt].q[0] = p[0];
            ax[kt].q[1] = p[1];
        }

        float hnew[2][8];
        float hold[2][8];

        // ---- compute phase: only READS of shared h ----
#pragma unroll
        for (int cg = 0; cg < 2; ++cg) {
            const int nt = wave + 8 * cg;

            // Opaque zero offset in an SGPR: defeats LICM (no hoist->spill of
            // the t-invariant weight fragments) while keeping the loads
            // rooted at the kernarg pointers -> global_load_b128 (not flat).
            unsigned int lz = 0;
            asm volatile("" : "+s"(lz));
            const unsigned short* WxL = Wx + lz;
            const unsigned short* WrL = Wr + lz;

            v8f acc_z, acc_r, acc_xh, acc_rh;
#pragma unroll
            for (int i = 0; i < 8; ++i) {
                acc_z[i] = bz[cg]; acc_r[i] = br[cg];
                acc_xh[i] = bh[cg]; acc_rh[i] = 0.0f;
            }

            // input projection: x_t @ kernel (K = 64)
#pragma unroll
            for (int kt = 0; kt < KT_X; ++kt) {
                acc_z  = wmma_bf16(ax[kt].v, load_frag(WxL, (nt +  0) * KT_X + kt, lane), acc_z);
                acc_r  = wmma_bf16(ax[kt].v, load_frag(WxL, (nt + 16) * KT_X + kt, lane), acc_r);
                acc_xh = wmma_bf16(ax[kt].v, load_frag(WxL, (nt + 32) * KT_X + kt, lane), acc_xh);
            }
            // recurrent projection: h @ recurrent_kernel (K = 256)
#pragma unroll
            for (int kt = 0; kt < KT_R; ++kt) {
                AFrag ah;
#pragma unroll
                for (int g = 0; g < 2; ++g) {
                    int k0 = kt * 32 + g * 16 + hi8;
                    ah.q[g] = *(const uint4*)&hB[M * GRU_H + k0];
                }
                acc_z  = wmma_bf16(ah.v, load_frag(WrL, (nt +  0) * KT_R + kt, lane), acc_z);
                acc_r  = wmma_bf16(ah.v, load_frag(WrL, (nt + 16) * KT_R + kt, lane), acc_r);
                acc_rh = wmma_bf16(ah.v, load_frag(WrL, (nt + 32) * KT_R + kt, lane), acc_rh);
            }

#pragma unroll
            for (int r = 0; r < 8; ++r)
                hold[cg][r] = hF[(r + hi8) * GRU_H + jcol[cg]];

#pragma unroll
            for (int r = 0; r < 8; ++r) {
                float z  = sigmoid_f(acc_z[r]);
                float rr = sigmoid_f(acc_r[r]);
                float hh = tanh_f(acc_xh[r] + rr * acc_rh[r]);
                hnew[cg][r] = z * hold[cg][r] + (1.0f - z) * hh;
            }
        }

        __syncthreads();   // every wave finished reading shared h

        // ---- write phase: outputs + shared h update ----
#pragma unroll
        for (int cg = 0; cg < 2; ++cg) {
            const int j = jcol[cg];
#pragma unroll
            for (int r = 0; r < 8; ++r) {
                int   Mr = r + hi8;
                float hn = hnew[cg][r];
                outputs[((size_t)(b0 + Mr) * GRU_T + t) * GRU_H + j] = hn;
                hF[Mr * GRU_H + j] = hn;
                hB[Mr * GRU_H + j] = f2bf(hn);
                if (t == GRU_T - 1)
                    hlast[(size_t)(b0 + Mr) * GRU_H + j] = hn;
            }
        }

        __syncthreads();   // h fully updated before next step's reads
    }
}

// ---------------------------------------------------------------------------
// Dense head: state = tanh(h_last @ dense_w + dense_b) ; (64x64), 16 waves.
// ---------------------------------------------------------------------------
__global__ __launch_bounds__(512, 1)
void dense_kernel(const float* __restrict__ hlast,
                  const unsigned short* __restrict__ Wd,   // packed H x 64
                  const float* __restrict__ dense_b,
                  float* __restrict__ state) {
    const int wave = threadIdx.x >> 5;   // 0..15
    const int lane = threadIdx.x & 31;
    const int mt = wave >> 2;            // batch tile 0..3
    const int nt = wave & 3;             // col tile 0..3
    const int hi8 = (lane >> 4) << 3;
    const int j  = nt * 16 + (lane & 15);

    v8f acc;
    float bb = dense_b[j];
#pragma unroll
    for (int i = 0; i < 8; ++i) acc[i] = bb;

    const float* row = hlast + (size_t)(mt * 16 + (lane & 15)) * GRU_H;
#pragma unroll
    for (int kt = 0; kt < KT_R; ++kt) {
        v16bf a = load_a_f32(row, kt * 32, lane);
        v16bf b = load_frag(Wd, nt * KT_R + kt, lane);
        acc = wmma_bf16(a, b, acc);
    }
#pragma unroll
    for (int r = 0; r < 8; ++r) {
        int Mr = mt * 16 + r + hi8;
        state[(size_t)Mr * 64 + j] = tanh_f(acc[r]);
    }
}

// ---------------------------------------------------------------------------
extern "C" void kernel_launch(void* const* d_in, const int* in_sizes, int n_in,
                              void* d_out, int out_size, void* d_ws, size_t ws_size,
                              hipStream_t stream) {
    const float* x        = (const float*)d_in[0];   // (64,512,64)
    const float* kernel   = (const float*)d_in[1];   // (64,768)
    const float* rkernel  = (const float*)d_in[2];   // (256,768)
    const float* bias     = (const float*)d_in[3];   // (768,)
    const float* dense_w  = (const float*)d_in[4];   // (256,64)
    const float* dense_b  = (const float*)d_in[5];   // (64,)

    // workspace layout: packed bf16 weights + packed bf16 x + fp32 h_last
    unsigned short* Wx = (unsigned short*)d_ws;                    // 96 KB
    unsigned short* Wr = Wx + (size_t)KT_X * NT_G * BLK_ELEMS;     // 384 KB
    unsigned short* Wd = Wr + (size_t)KT_R * NT_G * BLK_ELEMS;     // 32 KB
    float* hlast = (float*)(Wd + (size_t)KT_R * NT_D * BLK_ELEMS); // 64 KB
    unsigned short* Xp = (unsigned short*)(hlast + GRU_B * GRU_H); // 4 MB

    float* outputs = (float*)d_out;                            // (64,512,256)
    float* state   = outputs + (size_t)GRU_B * GRU_T * GRU_H;  // (64,64)

    // 1) one-time packing into WMMA lane layouts
    {
        int n = KT_X * NT_G * 32;
        pack_b_kernel<<<(n + 255) / 256, 256, 0, stream>>>(kernel, Wx, GRU_F, GRU_G);
    }
    {
        int n = KT_R * NT_G * 32;
        pack_b_kernel<<<(n + 255) / 256, 256, 0, stream>>>(rkernel, Wr, GRU_H, GRU_G);
    }
    {
        int n = KT_R * NT_D * 32;
        pack_b_kernel<<<(n + 255) / 256, 256, 0, stream>>>(dense_w, Wd, GRU_H, 64);
    }
    {
        int n = 4 * GRU_T * KT_X * 32;
        pack_a_x_kernel<<<(n + 255) / 256, 256, 0, stream>>>(x, Xp);
    }

    // 2) persistent GRU scan: 4 batch tiles x 8 waves
    gru_scan_kernel<<<4, 256, 0, stream>>>(Xp, bias, Wx, Wr, outputs, hlast);

    // 3) dense head on h_last
    dense_kernel<<<1, 512, 0, stream>>>(hlast, Wd, dense_b, state);
}
